// RD_SNNModel_Synaptic_44667659878464
// MI455X (gfx1250) — matrix operand order repaired and checked by reading, not compiled, not run on hardware
//
#include <hip/hip_runtime.h>
#include <hip/hip_bf16.h>

// ---------------------------------------------------------------------------
// SNN (snnTorch "Synaptic") forward, persistent-kernel formulation.
//   T=100, B=512, dims 700 -> 512 -> 512 -> 512 -> 512 -> 20
//   cur = x @ W^T + b ; syn = a*syn + cur ; mem = b*mem + syn - (mem_old>1)
//   spike = (mem > 1)
// One workgroup owns a 16-row batch tile and runs all layers & timesteps
// (rows are independent across the whole network -> no inter-block sync).
// GEMMs: v_wmma_f32_16x16x32_bf16, two N-tiles per pass sharing one A frag.
// ---------------------------------------------------------------------------

#define BATCH   512
#define NIN     700
#define NINP    704   // K padded to multiple of 32
#define NHID    512
#define NOUT    20
#define NOUTP   32    // N padded to multiple of 16

typedef unsigned short u16;
typedef __attribute__((ext_vector_type(4)))  u16    u16x4;
typedef __attribute__((ext_vector_type(8)))  u16    u16x8;
typedef __attribute__((ext_vector_type(16))) __bf16 v16bf;
typedef __attribute__((ext_vector_type(8)))  float  v8f;
typedef __attribute__((ext_vector_type(4)))  float  f32x4;

struct FragHalves { u16x8 lo, hi; };
union Frag { FragHalves h; v16bf v; };

// --- scratch layout (d_ws) --------------------------------------------------
// bf16 weights (padded), then fp32 syn state, then fp32 mem state.
constexpr size_t W0_OFF = 0;                          // 512 x 704
constexpr size_t W1_OFF = W0_OFF + (size_t)512 * 704; // 512 x 512
constexpr size_t W2_OFF = W1_OFF + (size_t)512 * 512;
constexpr size_t W3_OFF = W2_OFF + (size_t)512 * 512;
constexpr size_t W4_OFF = W3_OFF + (size_t)512 * 512; // 32 x 512 (rows padded)
constexpr size_t WTOT   = W4_OFF + (size_t)32 * 512;  // u16 elements
constexpr size_t STATE_OFF_BYTES = ((WTOT * 2 + 255) / 256) * 256;

// per-layer state offsets (floats): rows=512, doutPad = 512,512,512,512,32
constexpr size_t SYN_OFF0 = 0;
constexpr size_t SYN_OFF1 = SYN_OFF0 + (size_t)512 * 512;
constexpr size_t SYN_OFF2 = SYN_OFF1 + (size_t)512 * 512;
constexpr size_t SYN_OFF3 = SYN_OFF2 + (size_t)512 * 512;
constexpr size_t SYN_OFF4 = SYN_OFF3 + (size_t)512 * 512;
constexpr size_t SYN_TOT  = SYN_OFF4 + (size_t)512 * 32;

__device__ __forceinline__ u16 f2bf(float f) {
  unsigned int u = __float_as_uint(f);
  unsigned int r = (u + 0x7FFFu + ((u >> 16) & 1u)) >> 16;  // round-nearest-even
  return (u16)r;
}

// --- one-time scratch init: weight fp32->bf16 (padded) + zero state ---------
__global__ void init_ws(const float* __restrict__ W0, const float* __restrict__ W1,
                        const float* __restrict__ W2, const float* __restrict__ W3,
                        const float* __restrict__ W4, u16* __restrict__ wbf,
                        float* __restrict__ state, size_t stateFloats) {
  const size_t i0 = (size_t)blockIdx.x * blockDim.x + threadIdx.x;
  const size_t stride = (size_t)gridDim.x * blockDim.x;
  for (size_t i = i0; i < (size_t)512 * 704; i += stride) {
    int r = (int)(i / 704), c = (int)(i % 704);
    wbf[W0_OFF + i] = (c < NIN) ? f2bf(W0[(size_t)r * NIN + c]) : (u16)0;
  }
  for (size_t i = i0; i < (size_t)512 * 512; i += stride) {
    wbf[W1_OFF + i] = f2bf(W1[i]);
    wbf[W2_OFF + i] = f2bf(W2[i]);
    wbf[W3_OFF + i] = f2bf(W3[i]);
  }
  for (size_t i = i0; i < (size_t)32 * 512; i += stride) {
    int r = (int)(i >> 9);
    wbf[W4_OFF + i] = (r < NOUT) ? f2bf(W4[i]) : (u16)0;
  }
  for (size_t i = i0; i < stateFloats; i += stride) state[i] = 0.0f;
}

// --- synaptic state update + spike emit for one 16x16 D tile ----------------
template <int DOUT, int DOUTP, bool LAST>
__device__ __forceinline__ void epilogue(const v8f& acc, int nbase,
                                         const float* __restrict__ bias,
                                         float* __restrict__ syn,
                                         float* __restrict__ mem, u16* xout,
                                         float* __restrict__ outp, int rowbase,
                                         int m, int hi) {
  const int   n  = nbase + m;          // D tile: N = lane&15
  const float bn = bias[n];
#pragma unroll
  for (int r = 0; r < 8; ++r) {        // D tile: M = r + 8*hi
    const int mloc = r + 8 * hi;
    const int row  = rowbase + mloc;
    const size_t sidx = (size_t)row * DOUTP + n;
    const float cur = acc[r] + bn;
    const float so  = syn[sidx];
    const float mo  = mem[sidx];
    const float reset = (mo > 1.0f) ? 1.0f : 0.0f;     // spike(mem_old - THR)
    const float sn = 0.9f * so + cur;
    const float mn = 0.8f * mo + sn - reset;           // THR = 1.0
    syn[sidx] = sn;
    mem[sidx] = mn;
    const bool spk = mn > 1.0f;
    if constexpr (!LAST) {
      xout[mloc * NINP + n] = spk ? (u16)0x3F80 : (u16)0;  // bf16 1.0 / 0.0
    } else {
      if (n < DOUT) outp[(size_t)row * NOUT + n] = spk ? 1.0f : 0.0f;
    }
  }
}

// --- one fused layer step for a 16-row batch tile ---------------------------
// XK: K extent (== bf16 weight row pitch), DOUT real outputs, DOUTP padded.
// Two N-tiles per pass share one A fragment: 1 WMMA per 1 ds_load_b128.
template <int XK, int DOUT, int DOUTP, bool LAST>
__device__ __forceinline__ void layer_step(
    const u16* __restrict__ W, const float* __restrict__ bias,
    float* __restrict__ syn, float* __restrict__ mem,
    const u16* xin, u16* xout, float* __restrict__ outp,
    int rowbase, int lane, int wave) {
  constexpr int KSTEPS = XK / 32;
  constexpr int NPAIRS = (DOUTP / 16) / 2;
  const int m  = lane & 15;
  const int hi = lane >> 4;
  for (int pr = wave; pr < NPAIRS; pr += 8) {  // wave-uniform: EXEC stays full
    const int nbase0 = pr * 32;
    const int nbase1 = nbase0 + 16;
    v8f acc0 = {0.f, 0.f, 0.f, 0.f, 0.f, 0.f, 0.f, 0.f};
    v8f acc1 = acc0;
    // A (16x32 bf16): lane(m,hi) lo half = K 8*hi+0..7, hi half = K 8*hi+16..23
    const u16* arow  = xin + m * NINP + 8 * hi;
    // B (32x16 bf16): lane n=m lo half = K 8*hi+0..7, hi half = K 8*hi+16..23 of weight row nbase+n
    const u16* brow0 = W + (size_t)(nbase0 + m) * XK + 8 * hi;
    const u16* brow1 = W + (size_t)(nbase1 + m) * XK + 8 * hi;
#pragma unroll
    for (int k = 0; k < KSTEPS; ++k) {
      Frag a, b0, b1;
      a.h.lo  = *(const u16x8*)(arow + 32 * k);
      a.h.hi  = *(const u16x8*)(arow + 32 * k + 16);
      b0.h.lo = *(const u16x8*)(brow0 + 32 * k);
      b0.h.hi = *(const u16x8*)(brow0 + 32 * k + 16);
      b1.h.lo = *(const u16x8*)(brow1 + 32 * k);
      b1.h.hi = *(const u16x8*)(brow1 + 32 * k + 16);
      acc0 = __builtin_amdgcn_wmma_f32_16x16x32_bf16(
          false, a.v, false, b0.v, (short)0, acc0, false, false);
      acc1 = __builtin_amdgcn_wmma_f32_16x16x32_bf16(
          false, a.v, false, b1.v, (short)0, acc1, false, false);
    }
    epilogue<DOUT, DOUTP, LAST>(acc0, nbase0, bias, syn, mem, xout, outp,
                                rowbase, m, hi);
    epilogue<DOUT, DOUTP, LAST>(acc1, nbase1, bias, syn, mem, xout, outp,
                                rowbase, m, hi);
  }
}

// --- persistent kernel: one block = 16 batch rows, all layers, all timesteps
__global__ void __launch_bounds__(256, 1) snn_persistent(
    const float* __restrict__ data, const float* __restrict__ b0,
    const float* __restrict__ b1, const float* __restrict__ b2,
    const float* __restrict__ b3, const float* __restrict__ b4,
    const u16* __restrict__ wbf, float* __restrict__ state,
    float* __restrict__ out, const int* __restrict__ tsPtr) {
  __shared__ __align__(16) u16 xbuf[2][16 * NINP];  // 45 KB double buffer
  const int tid = threadIdx.x;
  const int wave = tid >> 5;
  const int lane = tid & 31;
  const int rowbase = blockIdx.x * 16;
  float* stateS = state;
  float* stateM = state + SYN_TOT;
  const int T = tsPtr[0];

  for (int t = 0; t < T; ++t) {
    // stage timestep input (fp32 -> bf16, zero-pad 700 -> 704), float4 chunks
    const float* src = data + ((size_t)t * BATCH + rowbase) * NIN;
    for (int i = tid; i < 16 * (NINP / 4); i += 256) {
      const int r = i / (NINP / 4);
      const int c = (i - r * (NINP / 4)) * 4;
      u16x4 v;
      if (c < NIN) {
        const f32x4 f = *(const f32x4*)(src + (size_t)r * NIN + c);
        v = (u16x4){f2bf(f.x), f2bf(f.y), f2bf(f.z), f2bf(f.w)};
      } else {
        v = (u16x4){0, 0, 0, 0};
      }
      *(u16x4*)(&xbuf[0][r * NINP + c]) = v;
    }
    __syncthreads();

    layer_step<NINP, NHID, NHID, false>(wbf + W0_OFF, b0, stateS + SYN_OFF0,
                                        stateM + SYN_OFF0, xbuf[0], xbuf[1],
                                        nullptr, rowbase, lane, wave);
    __syncthreads();
    layer_step<NHID, NHID, NHID, false>(wbf + W1_OFF, b1, stateS + SYN_OFF1,
                                        stateM + SYN_OFF1, xbuf[1], xbuf[0],
                                        nullptr, rowbase, lane, wave);
    __syncthreads();
    layer_step<NHID, NHID, NHID, false>(wbf + W2_OFF, b2, stateS + SYN_OFF2,
                                        stateM + SYN_OFF2, xbuf[0], xbuf[1],
                                        nullptr, rowbase, lane, wave);
    __syncthreads();
    layer_step<NHID, NHID, NHID, false>(wbf + W3_OFF, b3, stateS + SYN_OFF3,
                                        stateM + SYN_OFF3, xbuf[1], xbuf[0],
                                        nullptr, rowbase, lane, wave);
    __syncthreads();
    layer_step<NHID, NOUT, NOUTP, true>(wbf + W4_OFF, b4, stateS + SYN_OFF4,
                                        stateM + SYN_OFF4, xbuf[0], nullptr,
                                        out + (size_t)t * BATCH * NOUT,
                                        rowbase, lane, wave);
    __syncthreads();  // before next timestep overwrites xbuf[0]
  }
}

extern "C" void kernel_launch(void* const* d_in, const int* in_sizes, int n_in,
                              void* d_out, int out_size, void* d_ws, size_t ws_size,
                              hipStream_t stream) {
  (void)in_sizes; (void)n_in; (void)out_size; (void)ws_size;
  const float* data = (const float*)d_in[0];
  const float* W0 = (const float*)d_in[1];  const float* b0 = (const float*)d_in[2];
  const float* W1 = (const float*)d_in[3];  const float* b1 = (const float*)d_in[4];
  const float* W2 = (const float*)d_in[5];  const float* b2 = (const float*)d_in[6];
  const float* W3 = (const float*)d_in[7];  const float* b3 = (const float*)d_in[8];
  const float* W4 = (const float*)d_in[9];  const float* b4 = (const float*)d_in[10];
  const int*   ts = (const int*)d_in[11];

  u16*   wbf   = (u16*)d_ws;
  float* state = (float*)((char*)d_ws + STATE_OFF_BYTES);
  const size_t stateFloats = 2 * SYN_TOT;

  init_ws<<<512, 256, 0, stream>>>(W0, W1, W2, W3, W4, wbf, state, stateFloats);
  snn_persistent<<<BATCH / 16, 256, 0, stream>>>(data, b0, b1, b2, b3, b4, wbf,
                                                 state, (float*)d_out, ts);
}
